// SimpleStackedAttn_30021821399347
// MI455X (gfx1250) — matrix operand order repaired and checked by reading, not compiled, not run on hardware
//
#include <hip/hip_runtime.h>

#define B_    64
#define N_    512
#define DIN_  256
#define DH_   512
#define DOUT_ 256
#define H_    4
#define M_    (B_ * N_)   // 32768 rows

typedef __bf16 bf16;
typedef __attribute__((ext_vector_type(16))) __bf16 v16bf;
typedef __attribute__((ext_vector_type(8)))  __bf16 v8bf;
typedef __attribute__((ext_vector_type(8)))  float  v8f;

// ---------------------------------------------------------------------------
// WMMA helper: D = A(16x32 bf16) * B(32x16 bf16) + C(16x16 f32)
// ---------------------------------------------------------------------------
static __device__ __forceinline__ v8f wmma_bf16(v16bf a, v16bf b, v8f c) {
  return __builtin_amdgcn_wmma_f32_16x16x32_bf16(
      /*neg_a=*/false, a, /*neg_b=*/false, b,
      /*c_mod=*/(short)0, c, /*reuse_a=*/false, /*reuse_b=*/false);
}

// A-operand (16x32, 16-bit): lane L<16 holds row L, K={0..7,16..23};
// lane L+16 holds row L, K={8..15,24..31}.  Two 16-byte loads.
static __device__ __forceinline__ v16bf load_a_frag(const bf16* base, int stride, int lane) {
  int lr  = lane & 15;
  int sel = (lane >> 4) & 1;
  const bf16* p = base + (size_t)lr * stride + sel * 8;
  union { v16bf v; v8bf h[2]; } u;
  u.h[0] = *(const v8bf*)(p);
  u.h[1] = *(const v8bf*)(p + 16);
  return u.v;
}

// B-operand (32x16, 16-bit): lane L<16 holds column L, K={0..7,16..23};
// lane L+16 holds column L, K={8..15,24..31}.  Two 16-byte loads.
static __device__ __forceinline__ v16bf load_b_frag(const bf16* base, int stride, int lane) {
  int lc  = lane & 15;
  int sel = (lane >> 4) & 1;
  const bf16* p = base + (size_t)lc * stride + sel * 8;
  union { v16bf v; v8bf h[2]; } u;
  u.h[0] = *(const v8bf*)(p);
  u.h[1] = *(const v8bf*)(p + 16);
  return u.v;
}

// ---------------------------------------------------------------------------
// f32 -> bf16 elementwise convert
// ---------------------------------------------------------------------------
__global__ void k_cvt_f32_bf16(const float* __restrict__ in, bf16* __restrict__ out, long n) {
  long i      = (long)blockIdx.x * blockDim.x + threadIdx.x;
  long stride = (long)gridDim.x * blockDim.x;
  for (; i < n; i += stride) out[i] = (bf16)in[i];
}

// ---------------------------------------------------------------------------
// Weight transpose + convert: w[h][d][c] f32 -> wT[h][c][d] bf16
// ---------------------------------------------------------------------------
__global__ void k_wT(const float* __restrict__ w, bf16* __restrict__ wT,
                     int Hn, int Kd, int Nc) {
  long total = (long)Hn * Kd * Nc;
  for (long i = (long)blockIdx.x * blockDim.x + threadIdx.x; i < total;
       i += (long)gridDim.x * blockDim.x) {
    int  h   = (int)(i / ((long)Kd * Nc));
    long rem = i - (long)h * Kd * Nc;
    int  d   = (int)(rem / Nc);
    int  c   = (int)(rem % Nc);
    wT[((size_t)h * Nc + c) * Kd + d] = (bf16)w[i];
  }
}

// ---------------------------------------------------------------------------
// K/Q producer GEMM: out = x[M][Kd] @ wT[h][2*DH][Kd]^T + bias, split into
// K (cols 0..DH-1) and Q (cols DH..2DH-1), bf16 row-major [h][M][DH].
// Block = 128 threads (4 waves); wave computes a 32x64 tile; software
// pipelined and unrolled x2 so the ping-pong buffers live in distinct
// registers (no v_mov rotation, longer WAR distance after WMMAs).
// ---------------------------------------------------------------------------
__global__ __launch_bounds__(128)
void k_gemm_kq(const bf16* __restrict__ x, const bf16* __restrict__ wT,
               const float* __restrict__ bias, bf16* __restrict__ Ko,
               bf16* __restrict__ Qo, int Kd) {
  const int tid = threadIdx.x, wave = tid >> 5, lane = tid & 31;
  const int h    = blockIdx.z;
  const long row0 = (long)blockIdx.x * 32;
  const int col0 = (blockIdx.y * 4 + wave) * 64;
  const bf16* A0 = x + (size_t)row0 * Kd;
  const bf16* A1 = A0 + (size_t)16 * Kd;
  const bf16* W0 = wT + ((size_t)h * (2 * DH_) + col0) * Kd;

  v8f acc0[4] = {}, acc1[4] = {};
  v16bf a0 = load_a_frag(A0, Kd, lane);
  v16bf a1 = load_a_frag(A1, Kd, lane);
  v16bf bb[4];
#pragma unroll
  for (int t = 0; t < 4; ++t) bb[t] = load_b_frag(W0 + (size_t)t * 16 * Kd, Kd, lane);

#pragma unroll 2
  for (int d0 = 0; d0 < Kd; d0 += 32) {
    const int dn = (d0 + 32 < Kd) ? d0 + 32 : d0;   // last iter: benign reload
    v16bf an0 = load_a_frag(A0 + dn, Kd, lane);
    v16bf an1 = load_a_frag(A1 + dn, Kd, lane);
    v16bf bn[4];
#pragma unroll
    for (int t = 0; t < 4; ++t) bn[t] = load_b_frag(W0 + (size_t)t * 16 * Kd + dn, Kd, lane);
#pragma unroll
    for (int t = 0; t < 4; ++t) {
      acc0[t] = wmma_bf16(a0, bb[t], acc0[t]);
      acc1[t] = wmma_bf16(a1, bb[t], acc1[t]);
    }
    a0 = an0; a1 = an1;
#pragma unroll
    for (int t = 0; t < 4; ++t) bb[t] = bn[t];
  }

  const int col16 = lane & 15, rh = lane >> 4;
#pragma unroll
  for (int t = 0; t < 4; ++t) {
    int   c  = col0 + t * 16 + col16;
    float bv = bias[h * 2 * DH_ + c];
    bf16* o  = (c < DH_) ? Ko : Qo;
    int   oc = (c < DH_) ? c : c - DH_;
#pragma unroll
    for (int r = 0; r < 8; ++r) {
      long m0 = row0 + rh * 8 + r;
      o[((size_t)h * M_ + m0) * DH_ + oc]        = (bf16)(acc0[t][r] + bv);
      o[((size_t)h * M_ + m0 + 16) * DH_ + oc]   = (bf16)(acc1[t][r] + bv);
    }
  }
}

// ---------------------------------------------------------------------------
// Embed producer GEMM: emb = x @ e_w[h] + e_b[h], written TRANSPOSED per
// (h,b): embT[h][b][col][n] bf16 (so attention PV B-operand is contiguous).
// Same 32x64 pipelined wave tile.
// ---------------------------------------------------------------------------
__global__ __launch_bounds__(128)
void k_gemm_emb(const bf16* __restrict__ x, const bf16* __restrict__ wT,
                const float* __restrict__ bias, bf16* __restrict__ embT, int Kd) {
  const int tid = threadIdx.x, wave = tid >> 5, lane = tid & 31;
  const int h    = blockIdx.z;
  const long row0 = (long)blockIdx.x * 32;
  const int col0 = (blockIdx.y * 4 + wave) * 64;
  const bf16* A0 = x + (size_t)row0 * Kd;
  const bf16* A1 = A0 + (size_t)16 * Kd;
  const bf16* W0 = wT + ((size_t)h * DH_ + col0) * Kd;

  v8f acc0[4] = {}, acc1[4] = {};
  v16bf a0 = load_a_frag(A0, Kd, lane);
  v16bf a1 = load_a_frag(A1, Kd, lane);
  v16bf bb[4];
#pragma unroll
  for (int t = 0; t < 4; ++t) bb[t] = load_b_frag(W0 + (size_t)t * 16 * Kd, Kd, lane);

#pragma unroll 2
  for (int d0 = 0; d0 < Kd; d0 += 32) {
    const int dn = (d0 + 32 < Kd) ? d0 + 32 : d0;
    v16bf an0 = load_a_frag(A0 + dn, Kd, lane);
    v16bf an1 = load_a_frag(A1 + dn, Kd, lane);
    v16bf bn[4];
#pragma unroll
    for (int t = 0; t < 4; ++t) bn[t] = load_b_frag(W0 + (size_t)t * 16 * Kd + dn, Kd, lane);
#pragma unroll
    for (int t = 0; t < 4; ++t) {
      acc0[t] = wmma_bf16(a0, bb[t], acc0[t]);
      acc1[t] = wmma_bf16(a1, bb[t], acc1[t]);
    }
    a0 = an0; a1 = an1;
#pragma unroll
    for (int t = 0; t < 4; ++t) bb[t] = bn[t];
  }

  const int col16 = lane & 15, rh = lane >> 4;
  const int bb_i = (int)(row0 / N_);
  const int n0   = (int)(row0 - (long)bb_i * N_);
#pragma unroll
  for (int t = 0; t < 4; ++t) {
    int   c  = col0 + t * 16 + col16;
    float bv = bias[h * DH_ + c];
    size_t base = (((size_t)h * B_ + bb_i) * DH_ + c) * N_;
#pragma unroll
    for (int r = 0; r < 8; ++r) {
      embT[base + n0 + rh * 8 + r]      = (bf16)(acc0[t][r] + bv);
      embT[base + n0 + 16 + rh * 8 + r] = (bf16)(acc1[t][r] + bv);
    }
  }
}

// ---------------------------------------------------------------------------
// Attention: one workgroup (8 waves) per (b, q-tile of 16 queries).
// Heads looped inside; flash softmax (unscaled logits, per reference);
// output = mean over heads, f32 [M][DH].
// ---------------------------------------------------------------------------
__global__ __launch_bounds__(256)
void k_attn(const bf16* __restrict__ Q, const bf16* __restrict__ K,
            const bf16* __restrict__ embT, float* __restrict__ out) {
  const int tid = threadIdx.x, wave = tid >> 5, lane = tid & 31;
  const int lr = lane & 15, rh = lane >> 4;
  const int b = blockIdx.x, qt = blockIdx.y;
  const long qrow0 = (long)b * N_ + qt * 16;

  __shared__ float s_part[8][16][33];                 // per-wave partial S tiles
  __shared__ __align__(16) bf16 s_p[16][40];          // softmaxed P (stride 40)
  __shared__ float s_scale[16];                       // online-softmax rescale
  __shared__ float s_linv[16];                        // 1/l at head end

  const v8f vz = {};
  v8f om[4] = {};                                     // mean-over-heads acc

  for (int h = 0; h < H_; ++h) {
    const bf16* Qh = Q + ((size_t)h * M_ + qrow0) * DH_;
    const bf16* Kh = K + ((size_t)h * M_ + (long)b * N_) * DH_;
    const bf16* Eh = embT + ((size_t)h * B_ + b) * (size_t)DH_ * N_;

    // Q fragments for this wave's 64-wide d slice, kept in registers
    v16bf qf0 = load_a_frag(Qh + wave * 64, DH_, lane);
    v16bf qf1 = load_a_frag(Qh + wave * 64 + 32, DH_, lane);

    v8f oh[4]; oh[0] = vz; oh[1] = vz; oh[2] = vz; oh[3] = vz;
    float m_r = -3.0e38f, l_r = 0.0f;                 // wave0 lanes 0..15 own these

    for (int kb = 0; kb < N_; kb += 32) {
      // --- partial S = Q[:, dslice] * K[:, dslice]^T for 32 keys ------------
#pragma unroll
      for (int t = 0; t < 2; ++t) {
        const bf16* kbase = Kh + (size_t)(kb + t * 16) * DH_ + wave * 64;
        v8f s = vz;
        s = wmma_bf16(qf0, load_b_frag(kbase, DH_, lane), s);
        s = wmma_bf16(qf1, load_b_frag(kbase + 32, DH_, lane), s);
#pragma unroll
        for (int r = 0; r < 8; ++r) s_part[wave][rh * 8 + r][t * 16 + lr] = s[r];
      }
      __syncthreads();

      // --- wave0: reduce partials + online softmax --------------------------
      if (wave == 0 && lane < 16) {
        const int row = lane;
        float rowmax = -3.0e38f;
        for (int c = 0; c < 32; ++c) {
          float s = 0.0f;
#pragma unroll
          for (int w2 = 0; w2 < 8; ++w2) s += s_part[w2][row][c];
          s_part[0][row][c] = s;                      // stash combined logits
          rowmax = fmaxf(rowmax, s);
        }
        float mnew = fmaxf(m_r, rowmax);
        float fac  = __expf(m_r - mnew);
        float sum  = 0.0f;
        for (int c = 0; c < 32; ++c) {
          float e = __expf(s_part[0][row][c] - mnew);
          sum += e;
          s_p[row][c] = (bf16)e;
        }
        l_r = l_r * fac + sum;
        m_r = mnew;
        s_scale[row] = fac;
      }
      __syncthreads();

      // --- all waves: rescale O and accumulate P @ emb ----------------------
      v16bf pa = load_a_frag(&s_p[0][0], 40, lane);
      float fr[8];
#pragma unroll
      for (int r = 0; r < 8; ++r) fr[r] = s_scale[rh * 8 + r];
#pragma unroll
      for (int t = 0; t < 4; ++t) {
#pragma unroll
        for (int r = 0; r < 8; ++r) oh[t][r] *= fr[r];
        const bf16* eb = Eh + (size_t)(wave * 64 + t * 16) * N_ + kb;
        oh[t] = wmma_bf16(pa, load_b_frag(eb, N_, lane), oh[t]);
      }
      __syncthreads();
    }

    // --- finalize head: divide by l, add (1/H) into mean accumulator --------
    if (wave == 0 && lane < 16) s_linv[lane] = 1.0f / fmaxf(l_r, 1e-30f);
    __syncthreads();
#pragma unroll
    for (int t = 0; t < 4; ++t)
#pragma unroll
      for (int r = 0; r < 8; ++r)
        om[t][r] += 0.25f * s_linv[rh * 8 + r] * oh[t][r];
    __syncthreads();
  }

  // --- store mean output, f32 row-major [M][DH] -----------------------------
#pragma unroll
  for (int t = 0; t < 4; ++t) {
    int c = wave * 64 + t * 16 + lr;
#pragma unroll
    for (int r = 0; r < 8; ++r)
      out[(size_t)(qrow0 + rh * 8 + r) * DH_ + c] = om[t][r];
  }
}

// ---------------------------------------------------------------------------
// Final projection: out = h @ proj_w + proj_b   (f32 output [M][DOUT])
// 32x64 pipelined wave tile; 4 waves cover all 256 output columns.
// ---------------------------------------------------------------------------
__global__ __launch_bounds__(128)
void k_proj(const bf16* __restrict__ x, const bf16* __restrict__ wT,
            const float* __restrict__ bias, float* __restrict__ out) {
  const int tid = threadIdx.x, wave = tid >> 5, lane = tid & 31;
  const long row0 = (long)blockIdx.x * 32;
  const int col0 = wave * 64;
  const bf16* A0 = x + (size_t)row0 * DH_;
  const bf16* A1 = A0 + (size_t)16 * DH_;
  const bf16* W0 = wT + (size_t)col0 * DH_;

  v8f acc0[4] = {}, acc1[4] = {};
  v16bf a0 = load_a_frag(A0, DH_, lane);
  v16bf a1 = load_a_frag(A1, DH_, lane);
  v16bf bb[4];
#pragma unroll
  for (int t = 0; t < 4; ++t) bb[t] = load_b_frag(W0 + (size_t)t * 16 * DH_, DH_, lane);

#pragma unroll 2
  for (int d0 = 0; d0 < DH_; d0 += 32) {
    const int dn = (d0 + 32 < DH_) ? d0 + 32 : d0;
    v16bf an0 = load_a_frag(A0 + dn, DH_, lane);
    v16bf an1 = load_a_frag(A1 + dn, DH_, lane);
    v16bf bn[4];
#pragma unroll
    for (int t = 0; t < 4; ++t) bn[t] = load_b_frag(W0 + (size_t)t * 16 * DH_ + dn, DH_, lane);
#pragma unroll
    for (int t = 0; t < 4; ++t) {
      acc0[t] = wmma_bf16(a0, bb[t], acc0[t]);
      acc1[t] = wmma_bf16(a1, bb[t], acc1[t]);
    }
    a0 = an0; a1 = an1;
#pragma unroll
    for (int t = 0; t < 4; ++t) bb[t] = bn[t];
  }

  const int col16 = lane & 15, rh = lane >> 4;
#pragma unroll
  for (int t = 0; t < 4; ++t) {
    int   c  = col0 + t * 16 + col16;
    float bv = bias[c];
#pragma unroll
    for (int r = 0; r < 8; ++r) {
      out[(size_t)(row0 + rh * 8 + r) * DOUT_ + c]      = acc0[t][r] + bv;
      out[(size_t)(row0 + 16 + rh * 8 + r) * DOUT_ + c] = acc1[t][r] + bv;
    }
  }
}

// ---------------------------------------------------------------------------
// Host-side launch sequence
// ---------------------------------------------------------------------------
extern "C" void kernel_launch(void* const* d_in, const int* in_sizes, int n_in,
                              void* d_out, int out_size, void* d_ws, size_t ws_size,
                              hipStream_t stream) {
  (void)in_sizes; (void)n_in; (void)out_size; (void)ws_size;
  const float* x      = (const float*)d_in[0];
  const float* e_w0   = (const float*)d_in[1];
  const float* e_b0   = (const float*)d_in[2];
  const float* kq_w0  = (const float*)d_in[3];
  const float* kq_b0  = (const float*)d_in[4];
  const float* e_w1   = (const float*)d_in[5];
  const float* e_b1   = (const float*)d_in[6];
  const float* kq_w1  = (const float*)d_in[7];
  const float* kq_b1  = (const float*)d_in[8];
  const float* proj_w = (const float*)d_in[9];
  const float* proj_b = (const float*)d_in[10];
  float* out = (float*)d_out;

  char* ws = (char*)d_ws;
  size_t off = 0;
  auto alloc = [&](size_t bytes) -> void* {
    void* p = ws + off;
    off += (bytes + 255) & ~(size_t)255;
    return p;
  };

  bf16* xb     = (bf16*)alloc((size_t)M_ * DIN_ * 2);
  bf16* hb     = (bf16*)alloc((size_t)M_ * DH_ * 2);
  float* hf    = (float*)alloc((size_t)M_ * DH_ * 4);
  bf16* wT_e0  = (bf16*)alloc((size_t)H_ * DH_ * DIN_ * 2);
  bf16* wT_kq0 = (bf16*)alloc((size_t)H_ * 2 * DH_ * DIN_ * 2);
  bf16* wT_e1  = (bf16*)alloc((size_t)H_ * DH_ * DH_ * 2);
  bf16* wT_kq1 = (bf16*)alloc((size_t)H_ * 2 * DH_ * DH_ * 2);
  bf16* wT_pr  = (bf16*)alloc((size_t)DOUT_ * DH_ * 2);
  bf16* Kb     = (bf16*)alloc((size_t)H_ * M_ * DH_ * 2);
  bf16* Qb     = (bf16*)alloc((size_t)H_ * M_ * DH_ * 2);
  bf16* Eb     = (bf16*)alloc((size_t)H_ * B_ * DH_ * N_ * 2);

  // --- input / weight conversion -------------------------------------------
  k_cvt_f32_bf16<<<1024, 256, 0, stream>>>(x, xb, (long)M_ * DIN_);
  k_wT<<<512, 256, 0, stream>>>(e_w0,  wT_e0,  H_, DIN_, DH_);
  k_wT<<<512, 256, 0, stream>>>(kq_w0, wT_kq0, H_, DIN_, 2 * DH_);
  k_wT<<<512, 256, 0, stream>>>(e_w1,  wT_e1,  H_, DH_,  DH_);
  k_wT<<<512, 256, 0, stream>>>(kq_w1, wT_kq1, H_, DH_,  2 * DH_);
  k_wT<<<512, 256, 0, stream>>>(proj_w, wT_pr, 1,  DH_,  DOUT_);

  // --- block 0 ---------------------------------------------------------------
  k_gemm_kq <<<dim3(M_ / 32, 4, H_), 128, 0, stream>>>(xb, wT_kq0, kq_b0, Kb, Qb, DIN_);
  k_gemm_emb<<<dim3(M_ / 32, 2, H_), 128, 0, stream>>>(xb, wT_e0, e_b0, Eb, DIN_);
  k_attn    <<<dim3(B_, N_ / 16),    256, 0, stream>>>(Qb, Kb, Eb, hf);
  k_cvt_f32_bf16<<<1024, 256, 0, stream>>>(hf, hb, (long)M_ * DH_);

  // --- block 1 ---------------------------------------------------------------
  k_gemm_kq <<<dim3(M_ / 32, 4, H_), 128, 0, stream>>>(hb, wT_kq1, kq_b1, Kb, Qb, DH_);
  k_gemm_emb<<<dim3(M_ / 32, 2, H_), 128, 0, stream>>>(hb, wT_e1, e_b1, Eb, DH_);
  k_attn    <<<dim3(B_, N_ / 16),    256, 0, stream>>>(Qb, Kb, Eb, hf);
  k_cvt_f32_bf16<<<1024, 256, 0, stream>>>(hf, hb, (long)M_ * DH_);

  // --- output projection -----------------------------------------------------
  k_proj<<<dim3(M_ / 32, 1), 128, 0, stream>>>(hb, wT_pr, proj_b, out);
}
